// Graph2Seq_88012469829796
// MI455X (gfx1250) — matrix operand order repaired and checked by reading, not compiled, not run on hardware
//
#include <hip/hip_runtime.h>
#include <hip/hip_bf16.h>

#define BB 256
#define NN 100
#define DD 128
#define HH 8
#define DHH 16
#define LL 3
#define DEG 16
#define BN (BB*NN)
#define EE (BN*DEG)
#define NEGV (-1000000000.0f)

typedef __attribute__((ext_vector_type(16))) _Float16 v16h;
typedef __attribute__((ext_vector_type(8)))  float    v8f;
typedef __attribute__((ext_vector_type(4)))  unsigned int u32x4;
typedef __attribute__((ext_vector_type(8)))  unsigned int u32x8;

// ---------------------------------------------------------------- input proj
__global__ void k_input(const float* __restrict__ x, const float* __restrict__ Win,
                        const float* __restrict__ bin, float* __restrict__ h) {
  int id = blockIdx.x * blockDim.x + threadIdx.x;
  if (id >= BN * DD) return;
  int i = id >> 7, d = id & 127;
  h[id] = x[2*i] * Win[d] + x[2*i+1] * Win[DD + d] + bin[d];
}

// ------------------------------------------------ WMMA GEMM: C = A(BN x D) @ W(D x D)
// One wave computes one 16x16 tile; 4 waves/block cover 64 rows of A.
// The shared 128x16 W tile is DMA'd into LDS once per block by the Tensor
// Data Mover (tensor_load_to_lds) with 1-dword row padding (17-dword pitch,
// bank-conflict-free fragment reads). f32 data -> f16 fragments in-register,
// f32 accumulation via v_wmma_f32_16x16x32_f16.
#define WT_PITCH 17
__global__ void k_gemm_wmma(const float* __restrict__ A, const float* __restrict__ W,
                            float* __restrict__ C) {
  __shared__ float Wt[DD * WT_PITCH];   // 128 rows x 17 dwords (16 data + 1 pad)

  int lane = threadIdx.x & 31;
  int wave = threadIdx.x >> 5;
  int m0 = blockIdx.x * 64 + wave * 16;
  int n0 = blockIdx.y * 16;
  int hi = (lane >> 4) & 1;   // upper half-wave flag
  int ml = lane & 15;

  // LDS byte offset of Wt: low 32 bits of the generic pointer (ptrtoint also
  // CAPTURES Wt so the asm "memory" clobber is known to reach it).
  unsigned lds_base = (unsigned)(unsigned long long)(const void*)&Wt[0];

  if (threadIdx.x < 32) {
    // ---- Tensor DMA: load W[:, n0:n0+16] (tile 16 x 128, stride 128) to LDS
    unsigned long long gaddr = (unsigned long long)(const void*)(W + n0);
    u32x4 g0;
    g0[0] = 1u;                                   // count=1, user descriptor
    g0[1] = lds_base;                             // lds_addr (bytes)
    g0[2] = (unsigned int)(gaddr & 0xFFFFFFFFull);           // global_addr[31:0]
    g0[3] = ((unsigned int)(gaddr >> 32) & 0x01FFFFFFu)      // global_addr[56:32]
            | 0x80000000u;                        // type = 2 ("image")
    u32x8 g1;
    g1[0] = 0x00020000u    // data_size = 2 (4 bytes), workgroup_mask = 0
          | 0x00100000u    // pad_enable = 1
          | 0x00C00000u;   // pad_interval = 3 (every 16 dwords), pad_amount = 0 (1 dword)
    g1[1] = (DD & 0xFFFFu) << 16;       // tensor_dim0 = 128 (low 16 in [63:48])
    g1[2] = (DD & 0xFFFFu) << 16;       // tensor_dim0 hi = 0; tensor_dim1 = 128 (low16)
    g1[3] = (16u) << 16;                // tensor_dim1 hi = 0; tile_dim0 = 16
    g1[4] = DD;                         // tile_dim1 = 128 (tile_dim2 = 0)
    g1[5] = DD;                         // tensor_dim0_stride = 128 (low 32)
    g1[6] = 0u;                         // stride hi = 0; tensor_dim1_stride = 0
    g1[7] = 0u;
    asm volatile("tensor_load_to_lds %0, %1" : : "s"(g0), "s"(g1) : "memory");
    __builtin_amdgcn_s_wait_tensorcnt(0);
  }
  __syncthreads();   // publish the W tile to all 4 waves

  v8f acc = {};
  for (int k0 = 0; k0 < DD; k0 += 32) {
    v16h a, b;
    // A fragment: 16x32 f16. K = j + 16*hi (contiguous, matches B)
    const float* arow = A + (size_t)(m0 + ml) * DD + k0 + hi * 16;
#pragma unroll
    for (int j = 0; j < 8; ++j) {
      a[j]     = (_Float16)arow[j];
      a[j + 8] = (_Float16)arow[8 + j];
    }
    // B fragment from LDS tile: K = k0 + j + 16*hi, N = ml
    const float* brow = Wt + (size_t)(k0 + hi * 16) * WT_PITCH + ml;
#pragma unroll
    for (int j = 0; j < 16; ++j)
      b[j] = (_Float16)brow[j * WT_PITCH];

    acc = __builtin_amdgcn_wmma_f32_16x16x32_f16(
        /*neg_a=*/false, a, /*neg_b=*/false, b,
        /*c_mod=*/(short)0, acc, /*reuse_a=*/false, /*reuse_b=*/false);
  }
  // C/D layout: VGPR r -> M = r + 8*hi, N = ml
#pragma unroll
  for (int r = 0; r < 8; ++r)
    C[(size_t)(m0 + r + hi * 8) * DD + n0 + ml] = acc[r];
}

// ------------------------------------- per-(node,head) attention coefficients
__global__ void k_attn_coef(const float* __restrict__ hh, const float* __restrict__ asrc,
                            const float* __restrict__ adst,
                            float* __restrict__ es, float* __restrict__ ed) {
  int id = blockIdx.x * blockDim.x + threadIdx.x;   // BN*H
  if (id >= BN * HH) return;
  int i = id >> 3, hd = id & 7;
  const float* v  = hh + (size_t)i * DD + hd * DHH;
  const float* as = asrc + hd * DHH;
  const float* ad = adst + hd * DHH;
  float s = 0.f, d2 = 0.f;
  for (int t = 0; t < DHH; ++t) { s += v[t] * as[t]; d2 += v[t] * ad[t]; }
  es[id] = s; ed[id] = d2;
}

// ---------------------- per-node GAT aggregate (exactly 16 in-edges per node)
__global__ void k_gat_agg(const float* __restrict__ hh, const int* __restrict__ src,
                          const float* __restrict__ eattr, const float* __restrict__ we,
                          const float* __restrict__ es, const float* __restrict__ ed,
                          float* __restrict__ h) {
  __shared__ float lg[HH * DEG];
  __shared__ float ex[HH * DEG];
  __shared__ float alpha[HH * DEG];
  __shared__ int   sidx[DEG];
  int i = blockIdx.x;
  int t = threadIdx.x;               // 128 = 8 heads x 16 nbrs
  int hd = t >> 4, j = t & 15;
  int e = i * DEG + j;
  if (t < DEG) sidx[t] = src[i * DEG + t];
  int s = src[e];
  float lv = es[s * HH + hd] + ed[i * HH + hd] + eattr[e] * we[hd];
  lg[t] = lv >= 0.f ? lv : 0.01f * lv;      // leaky_relu(0.01)
  __syncthreads();
  float m = -1e30f;
  for (int jj = 0; jj < DEG; ++jj) m = fmaxf(m, lg[hd * DEG + jj]);
  ex[t] = expf(lg[t] - m);
  __syncthreads();
  float den = 0.f;
  for (int jj = 0; jj < DEG; ++jj) den += ex[hd * DEG + jj];
  alpha[t] = ex[t] / (den + 1e-9f);
  __syncthreads();
  // message for feature d = t (head = t>>4), then ELU + residual (in-place safe)
  float acc = 0.f;
  for (int jj = 0; jj < DEG; ++jj)
    acc += alpha[hd * DEG + jj] * hh[(size_t)sidx[jj] * DD + t];
  float eluv = acc > 0.f ? acc : (expf(acc) - 1.f);
  h[(size_t)i * DD + t] += eluv;
}

// ----------------------------------------------- graph mean + start embedding
__global__ void k_pool(const float* __restrict__ h, float* __restrict__ gemb,
                       float* __restrict__ semb) {
  int id = blockIdx.x * blockDim.x + threadIdx.x;   // B*D
  if (id >= BB * DD) return;
  int b = id >> 7, d = id & 127;
  float s = 0.f;
  for (int n = 0; n < NN; ++n) s += h[(size_t)(b * NN + n) * DD + d];
  gemb[id] = s * (1.0f / NN);
  semb[id] = h[(size_t)(b * NN) * DD + d];
}

// -------------------- persistent decoder: one workgroup per batch row, 100 steps
__global__ void k_decode(const float* __restrict__ h,
                         const float* __restrict__ kbuf, const float* __restrict__ vbuf,
                         const float* __restrict__ k2buf,
                         const float* __restrict__ gemb, const float* __restrict__ semb,
                         const float* __restrict__ Wq, const float* __restrict__ Wo,
                         const float* __restrict__ tok1, const float* __restrict__ tokf,
                         float* __restrict__ out) {
  __shared__ float ctx[3 * DD];
  __shared__ float qs[DD];
  __shared__ float sc[HH * NN];
  __shared__ float gl[DD];
  __shared__ float g2[DD];
  __shared__ float lgt[NN];
  __shared__ float mask[NN];
  __shared__ float red[2];
  __shared__ int   selIdx;

  int b = blockIdx.x;
  int t = threadIdx.x;   // 128
  ctx[t]          = gemb[b * DD + t];
  ctx[DD + t]     = tok1[t];
  ctx[2 * DD + t] = tokf[t];
  if (t < NN) mask[t] = 0.f;
  __syncthreads();

  float* probs = out + (size_t)b * NN * NN;
  float* tours = out + (size_t)BB * NN * NN + (size_t)b * NN;
  const float* kb  = kbuf  + (size_t)b * NN * DD;
  const float* vb  = vbuf  + (size_t)b * NN * DD;
  const float* k2b = k2buf + (size_t)b * NN * DD;
  const float* hb  = h     + (size_t)b * NN * DD;
  const float rsdh = 0.25f;                   // 1/sqrt(16)
  const float rsd  = 0.08838834764831845f;    // 1/sqrt(128)

  for (int step = 0; step < NN; ++step) {
    // q = ctx(384) @ Wq(384x128)
    float acc = 0.f;
    for (int i = 0; i < 3 * DD; ++i) acc += ctx[i] * Wq[i * DD + t];
    qs[t] = acc;
    __syncthreads();
    // scores[h][n] = (q_h . k[b,n,h,:]) / sqrt(DH), masked
    for (int job = t; job < HH * NN; job += 128) {
      int hd = job / NN, n = job - hd * NN;
      const float* kr = kb + (size_t)n * DD + hd * DHH;
      const float* qr = qs + hd * DHH;
      float s = 0.f;
      for (int d2 = 0; d2 < DHH; ++d2) s += qr[d2] * kr[d2];
      s *= rsdh;
      if (mask[n] > 0.f) s = NEGV;
      sc[job] = s;
    }
    __syncthreads();
    // per-head softmax over N
    if (t < HH) {
      float m = -1e30f;
      for (int n = 0; n < NN; ++n) m = fmaxf(m, sc[t * NN + n]);
      float den = 0.f;
      for (int n = 0; n < NN; ++n) { float e2 = expf(sc[t * NN + n] - m); sc[t * NN + n] = e2; den += e2; }
      float inv = 1.f / den;
      for (int n = 0; n < NN; ++n) sc[t * NN + n] *= inv;
    }
    __syncthreads();
    // glimpse[d] = sum_n attn[h(d)][n] * v[b,n,d]
    {
      int hd = t >> 4;
      float a2 = 0.f;
      for (int n = 0; n < NN; ++n) a2 += sc[hd * NN + n] * vb[(size_t)n * DD + t];
      gl[t] = a2;
    }
    __syncthreads();
    // glimpse @ Wo
    {
      float a2 = 0.f;
      for (int i = 0; i < DD; ++i) a2 += gl[i] * Wo[i * DD + t];
      g2[t] = a2;
    }
    __syncthreads();
    // logits[n] = 10 * tanh((g2 . k2[b,n,:]) / sqrt(D)), masked
    if (t < NN) {
      float a2 = 0.f;
      for (int d2 = 0; d2 < DD; ++d2) a2 += g2[d2] * k2b[(size_t)t * DD + d2];
      float lv = 10.f * tanhf(a2 * rsd);
      if (mask[t] > 0.f) lv = NEGV;
      lgt[t] = lv;
    }
    __syncthreads();
    if (t == 0) {
      float m = lgt[0]; int idx = 0;
      for (int n = 1; n < NN; ++n) if (lgt[n] > m) { m = lgt[n]; idx = n; }  // first-max
      float den = 0.f;
      for (int n = 0; n < NN; ++n) den += expf(lgt[n] - m);
      red[0] = m; red[1] = den; selIdx = idx;
      tours[step] = (float)idx;
      mask[idx] = 1.f;
    }
    __syncthreads();
    if (t < NN) probs[(size_t)step * NN + t] = expf(lgt[t] - red[0]) / red[1];
    int idx = selIdx;
    // context = [graph_emb (already in place), start_emb, selected node]
    ctx[DD + t]     = semb[b * DD + t];
    ctx[2 * DD + t] = hb[(size_t)idx * DD + t];
    __syncthreads();
  }
}

// ---------------------------------------------------------------------------
extern "C" void kernel_launch(void* const* d_in, const int* in_sizes, int n_in,
                              void* d_out, int out_size, void* d_ws, size_t ws_size,
                              hipStream_t stream) {
  (void)in_sizes; (void)n_in; (void)out_size; (void)ws_size;
  const float* x      = (const float*)d_in[0];
  const int*   eidx   = (const int*)  d_in[1];   // [0:E) = src, [E:2E) = dst (implicit)
  const float* eattr  = (const float*)d_in[2];
  const float* W_in   = (const float*)d_in[3];
  const float* b_in   = (const float*)d_in[4];
  const float* W      = (const float*)d_in[5];   // (L, D, D)
  const float* a_src  = (const float*)d_in[6];   // (L, H, DH)
  const float* a_dst  = (const float*)d_in[7];
  const float* w_e    = (const float*)d_in[8];   // (L, H)
  const float* Wq     = (const float*)d_in[9];   // (3D, D)
  const float* Wk     = (const float*)d_in[10];
  const float* Wv     = (const float*)d_in[11];
  const float* Wo     = (const float*)d_in[12];
  const float* Wk2    = (const float*)d_in[13];
  const float* tok1   = (const float*)d_in[14];
  const float* tokf   = (const float*)d_in[15];
  float* out = (float*)d_out;

  const int* src = eidx;  // edge_index[0]

  // workspace layout (floats)
  float* ws   = (float*)d_ws;
  float* h    = ws;                              // BN*D
  float* hh   = h    + (size_t)BN * DD;          // BN*D
  float* kb   = hh   + (size_t)BN * DD;          // BN*D
  float* vb   = kb   + (size_t)BN * DD;          // BN*D
  float* k2b  = vb   + (size_t)BN * DD;          // BN*D
  float* es   = k2b  + (size_t)BN * DD;          // BN*H
  float* ed   = es   + (size_t)BN * HH;          // BN*H
  float* gemb = ed   + (size_t)BN * HH;          // B*D
  float* semb = gemb + (size_t)BB * DD;          // B*D

  dim3 blk(128);
  dim3 gemmGrid(BN / 64, DD / 16);

  // h0 = x @ W_in + b_in
  k_input<<<dim3((BN * DD + 127) / 128), blk, 0, stream>>>(x, W_in, b_in, h);

  // 3 GAT layers
  for (int l = 0; l < LL; ++l) {
    k_gemm_wmma<<<gemmGrid, blk, 0, stream>>>(h, W + (size_t)l * DD * DD, hh);
    k_attn_coef<<<dim3((BN * HH + 127) / 128), blk, 0, stream>>>(
        hh, a_src + l * HH * DHH, a_dst + l * HH * DHH, es, ed);
    k_gat_agg<<<dim3(BN), blk, 0, stream>>>(hh, src, eattr, w_e + l * HH, es, ed, h);
  }

  // pooling
  k_pool<<<dim3((BB * DD + 127) / 128), blk, 0, stream>>>(h, gemb, semb);

  // k, v, k2 projections
  k_gemm_wmma<<<gemmGrid, blk, 0, stream>>>(h, Wk,  kb);
  k_gemm_wmma<<<gemmGrid, blk, 0, stream>>>(h, Wv,  vb);
  k_gemm_wmma<<<gemmGrid, blk, 0, stream>>>(h, Wk2, k2b);

  // persistent decoder: one workgroup per batch row runs all 100 steps
  k_decode<<<dim3(BB), blk, 0, stream>>>(h, kb, vb, k2b, gemb, semb,
                                         Wq, Wo, tok1, tokf, out);
}
